// FuzzyPatternTM_STCM_21689584844984
// MI455X (gfx1250) — matrix operand order repaired and checked by reading, not compiled, not run on hardware
//
#include <hip/hip_runtime.h>
#include <hip/hip_bf16.h>

typedef __attribute__((ext_vector_type(16))) _Float16 v16h;
typedef __attribute__((ext_vector_type(8)))  _Float16 v8h;
typedef __attribute__((ext_vector_type(8)))  float    v8f;

#define FDIM 2048
#define CDIM 128
#define KOUT 10
#define NK   (FDIM / 32)          // 64 K-steps
#define WPAD 40                   // padded row stride (halfs) in LDS W buffer
#define PRODUCT_SCALE 0.001953125f   // 4/2048

// ---------------------------------------------------------------------------
// Kernel 1: quantize ternary masks to f16 weight matrix W[c][f] = -sign(logit)
// and base[c] = count(logit > 0). Rows 0..63 from pos_logits, 64..127 from neg.
// ---------------------------------------------------------------------------
__global__ void prep_masks(const float* __restrict__ pos,
                           const float* __restrict__ neg,
                           _Float16* __restrict__ W,
                           float* __restrict__ base) {
    const int c   = blockIdx.x;          // 0..127
    const int tid = threadIdx.x;         // 0..255
    const float* src = (c < 64) ? (pos + (size_t)c * FDIM)
                                : (neg + (size_t)(c - 64) * FDIM);
    float cnt = 0.0f;
    for (int f = tid; f < FDIM; f += 256) {
        float v = src[f];
        float w = (v > 0.0f) ? -1.0f : ((v < 0.0f) ? 1.0f : 0.0f);
        W[(size_t)c * FDIM + f] = (_Float16)w;
        cnt += (v > 0.0f) ? 1.0f : 0.0f;
    }
    __shared__ float red[256];
    red[tid] = cnt;
    __syncthreads();
    for (int s = 128; s > 0; s >>= 1) {
        if (tid < s) red[tid] += red[tid + s];
        __syncthreads();
    }
    if (tid == 0) base[c] = red[0];
}

// ---------------------------------------------------------------------------
// Async global -> LDS copy of one W K-chunk (128 rows x 32 f16 = 8 KB),
// padded to WPAD halfs per row. 512 b128 pieces, 2 per thread. ASYNCcnt.
// ---------------------------------------------------------------------------
__device__ __forceinline__ void async_copy_wchunk(const _Float16* __restrict__ Wg,
                                                  int kb, _Float16* buf, int tid) {
#pragma unroll
    for (int rep = 0; rep < 2; ++rep) {
        const int piece = tid + rep * 256;     // 0..511
        const int row   = piece >> 2;          // clause 0..127
        const int sub   = piece & 3;           // 16-byte piece within 64B row
        const unsigned lds_off =
            (unsigned)(uintptr_t)(buf + row * WPAD) + (unsigned)(sub * 16);
        const char* g = (const char*)(Wg + (size_t)row * FDIM + kb) + sub * 16;
        asm volatile("global_load_async_to_lds_b128 %0, %1, off"
                     :: "v"(lds_off), "v"(g) : "memory");
    }
}

// ---------------------------------------------------------------------------
// Kernel 2: fused GEMM (x @ W^T) with WMMA f16 -> clause strengths -> voting.
// Block: 256 threads = 8 waves; block tile = 64 rows x 128 clauses.
// Wave w: rows (w%4)*16, clause n-tiles (w/4)*4 .. +3  (16x64 per wave).
// W K-chunks double-buffered in LDS via async-to-LDS; B frags via ds_load.
// ---------------------------------------------------------------------------
__global__ void __launch_bounds__(256)
fused_tm(const float* __restrict__ x,
         const _Float16* __restrict__ W,
         const float* __restrict__ base,
         const float* __restrict__ cbias,
         const float* __restrict__ voting,
         float* __restrict__ out) {
    __shared__ alignas(16) _Float16 wbuf[2][CDIM * WPAD];  // 2 x 10 KB
    __shared__ float sv[64][132];        // clause votes tile (padded stride)
    __shared__ float svote[CDIM * KOUT]; // voting matrix cache

    const int tid = threadIdx.x;
    for (int i = tid; i < CDIM * KOUT; i += 256) svote[i] = voting[i];

    const int wave   = tid >> 5;
    const int lane   = tid & 31;
    const int rtile  = wave & 3;         // row tile within block
    const int cgrp   = wave >> 2;        // 0: clauses 0..63, 1: clauses 64..127
    const int lane16 = lane & 15;
    const int khalf  = lane >> 4;        // which K-half this lane holds

    const int bm = blockIdx.x * 64;      // block row base
    const int m0 = bm + rtile * 16;      // wave row base
    const float* xrow = x + (size_t)(m0 + lane16) * FDIM;

    v8f acc[4] = {};

    // Prologue: start async fill of chunk 0.
    async_copy_wchunk(W, 0, &wbuf[0][0], tid);

    for (int kt = 0; kt < NK; ++kt) {
        const int kb = kt * 32;

        // Chunk kt is in flight: wait for our pieces, then sync the group so
        // every piece (written by every thread) is visible, and so the buffer
        // we are about to overwrite (kt+1) has been fully consumed (at kt-1).
        asm volatile("s_wait_asynccnt 0x0" ::: "memory");
        __syncthreads();
        if (kt + 1 < NK)
            async_copy_wchunk(W, kb + 32, &wbuf[(kt + 1) & 1][0], tid);

        // ---- A fragment: 16x32 f16, ISA layout:
        // lanes 0-15 : K = kb+{0..7, 16..23};  lanes 16-31: K = kb+{8..15, 24..31}
        const float* p0 = xrow + kb + khalf * 8;
        float4 u0 = *(const float4*)(p0);
        float4 u1 = *(const float4*)(p0 + 4);
        float4 u2 = *(const float4*)(p0 + 16);
        float4 u3 = *(const float4*)(p0 + 20);
        v16h a;
        a[0]  = (_Float16)u0.x;  a[1]  = (_Float16)u0.y;
        a[2]  = (_Float16)u0.z;  a[3]  = (_Float16)u0.w;
        a[4]  = (_Float16)u1.x;  a[5]  = (_Float16)u1.y;
        a[6]  = (_Float16)u1.z;  a[7]  = (_Float16)u1.w;
        a[8]  = (_Float16)u2.x;  a[9]  = (_Float16)u2.y;
        a[10] = (_Float16)u2.z;  a[11] = (_Float16)u2.w;
        a[12] = (_Float16)u3.x;  a[13] = (_Float16)u3.y;
        a[14] = (_Float16)u3.z;  a[15] = (_Float16)u3.w;

        // ---- B fragments from LDS: 32x16 f16 per n-tile.
        // lanes 0-15 hold K = kb+{0..7, 16..23} of column n; lanes 16-31 hold K = kb+{8..15, 24..31}.
        const _Float16* bb = &wbuf[kt & 1][0];
#pragma unroll
        for (int j = 0; j < 4; ++j) {
            const int n = (cgrp * 4 + j) * 16 + lane16;
            const _Float16* wp = bb + n * WPAD + khalf * 8;
            v8h b0 = *(const v8h*)(wp);
            v8h b1 = *(const v8h*)(wp + 16);
            v16h b;
#pragma unroll
            for (int e = 0; e < 8; ++e) { b[e] = b0[e]; b[8 + e] = b1[e]; }
            acc[j] = __builtin_amdgcn_wmma_f32_16x16x32_f16(
                false, a, false, b, (short)0, acc[j], false, false);
        }
    }

    // ---- Epilogue: penalties -> strengths -> signed votes into LDS.
    // C/D layout: lanes 0-15: VGPR v -> M=v; lanes 16-31: VGPR v -> M=v+8; N=lane16.
    const int rbase = rtile * 16 + khalf * 8;
#pragma unroll
    for (int j = 0; j < 4; ++j) {
        const int n  = (cgrp * 4 + j) * 16 + lane16;
        const float bs = base[n];
        const float cb = cbias[n];
        const float sg = (n < 64) ? 1.0f : -1.0f;
#pragma unroll
        for (int v = 0; v < 8; ++v) {
            float pen = (bs + acc[j][v]) * PRODUCT_SCALE;
            pen = fminf(fmaxf(pen, 0.0f), 10.0f);
            sv[rbase + v][n] = sg * __expf(-pen) + cb;
        }
    }
    __syncthreads();

    // ---- Final contraction: [64 x 128] @ [128 x 10]
    for (int i = tid; i < 64 * KOUT; i += 256) {
        const int r = i / KOUT;
        const int k = i - r * KOUT;
        float sum = 0.0f;
#pragma unroll 8
        for (int n = 0; n < CDIM; ++n) sum += sv[r][n] * svote[n * KOUT + k];
        out[(size_t)(bm + r) * KOUT + k] = sum;
    }
}

// ---------------------------------------------------------------------------
extern "C" void kernel_launch(void* const* d_in, const int* in_sizes, int n_in,
                              void* d_out, int out_size, void* d_ws, size_t ws_size,
                              hipStream_t stream) {
    const float* x      = (const float*)d_in[0];   // [B, 2048]
    const float* pos    = (const float*)d_in[1];   // [64, 2048]
    const float* neg    = (const float*)d_in[2];   // [64, 2048]
    const float* cbias  = (const float*)d_in[3];   // [128]
    const float* voting = (const float*)d_in[4];   // [128, 10]
    float* out = (float*)d_out;

    _Float16* W   = (_Float16*)d_ws;                         // 128*2048*2 = 512 KB
    float*    bse = (float*)((char*)d_ws + (size_t)CDIM * FDIM * sizeof(_Float16));

    const int B = in_sizes[0] / FDIM;  // 16384

    prep_masks<<<CDIM, 256, 0, stream>>>(pos, neg, W, bse);
    fused_tm<<<B / 64, 256, 0, stream>>>(x, W, bse, cbias, voting, out);
}
